// MultiHeadedAttention_6227702579206
// MI455X (gfx1250) — matrix operand order repaired and checked by reading, not compiled, not run on hardware
//
#include <hip/hip_runtime.h>
#include <hip/hip_bf16.h>

// ---------------------------------------------------------------------------
// MultiHeadedAttention for MI455X (gfx1250, wave32, WMMA).
//   HIDDEN=768, HEADS=12, HEAD_DIM=64, B=2, S=4096.
// Pipeline:
//   gemm<0> x3 : X @ W + b  -> f16 [b,h,s,d]   (Q pre-scaled by 1/sqrt(64))
//   attn       : flash-attention, v_wmma_f32_16x16x32_f16 for QK^T and PV
//   gemm<1>    : ctx @ Wo + bo -> fp32 d_out
// ---------------------------------------------------------------------------

typedef _Float16 h16 __attribute__((ext_vector_type(16)));
typedef _Float16 h8  __attribute__((ext_vector_type(8)));
typedef float    f8  __attribute__((ext_vector_type(8)));

#define HIDDEN   768
#define HEADS    12
#define SEQ      4096
#define BATCH    2
#define ROWS     (BATCH*SEQ)      // 8192
#define LDG      40               // 32 + 8 pad (f16), keeps 16B alignment
#define LDV      72               // 64 + 8 pad (f16), keeps 16B alignment

__device__ __forceinline__ h16 cat8(h8 lo, h8 hi) {
  return __builtin_shufflevector(lo, hi, 0,1,2,3,4,5,6,7,8,9,10,11,12,13,14,15);
}

__device__ __forceinline__ f8 wmma_f16(h16 a, h16 b, f8 c) {
  // (neg_a, A, neg_b, B, c_mod, C, reuse_a, reuse_b)
  return __builtin_amdgcn_wmma_f32_16x16x32_f16(false, a, false, b, (short)0, c,
                                                false, false);
}

__device__ __forceinline__ h8 pack8(float4 a, float4 b) {
  h8 r;
  r[0]=(_Float16)a.x; r[1]=(_Float16)a.y; r[2]=(_Float16)a.z; r[3]=(_Float16)a.w;
  r[4]=(_Float16)b.x; r[5]=(_Float16)b.y; r[6]=(_Float16)b.z; r[7]=(_Float16)b.w;
  return r;
}

// A-fragment (16x32 f16, M x K): lane m = lane%16;
//   lanes 0-15 hold K=0..7 and 16..23, lanes 16-31 hold K=8..15 and 24..31.
__device__ __forceinline__ h16 ldsA(const _Float16* base, int mBase, int lane) {
  int ln = lane & 15, g = lane >> 4;
  const _Float16* p = base + (mBase + ln) * LDG + 8 * g;
  return cat8(*(const h8*)p, *(const h8*)(p + 16));
}

// B-fragment (32x16 f16, K x N) read from a [n][k] tile: lane n = lane%16;
//   lanes 0-15 hold K=0..7 and 16..23, lanes 16-31 hold K=8..15 and 24..31.
__device__ __forceinline__ h16 ldsB(const _Float16* base, int nBase, int lane) {
  int ln = lane & 15, g = lane >> 4;
  const _Float16* p = base + (nBase + ln) * LDG + 8 * g;
  return cat8(*(const h8*)p, *(const h8*)(p + 16));
}

// ---------------------------------------------------------------------------
// GEMM: [8192 x 768] (fp32) @ [768 x 768] (fp32) + bias.
// MODE 0: out f16 in [b, head, s, d] layout, value scaled by `scale`.
// MODE 1: out fp32 row-major [8192 x 768].
// Block 256 threads (8 waves); tile 128x128, K-step 32; wave tile 32x64.
// ---------------------------------------------------------------------------
template <int MODE>
__global__ __launch_bounds__(256) void gemm_kernel(
    const float* __restrict__ X, const float* __restrict__ W,
    const float* __restrict__ bias, void* __restrict__ outp, float scale) {
  __shared__ _Float16 As[128 * LDG];   // [m][k]
  __shared__ _Float16 Bs[128 * LDG];   // [n][k] (transposed during staging)

  const int tid  = threadIdx.x;
  const int wave = tid >> 5, lane = tid & 31;
  const int g = lane >> 4, ln = lane & 15;
  const int waveM = (wave & 3) * 32;    // 0,32,64,96
  const int waveN = (wave >> 2) * 64;   // 0,64
  const int blockM = blockIdx.y * 128;
  const int blockN = blockIdx.x * 128;

  const int arow = tid >> 1, acs = (tid & 1) * 16;   // A staging: 2 thr/row
  const int bkr  = tid >> 3, bns = (tid & 7) * 16;   // B staging: 8 thr/row

  f8 acc[2][4] = {};

  for (int k0 = 0; k0 < HIDDEN; k0 += 32) {
    // ---- stage A tile: 128x32 fp32 -> f16, row-major -------------------
    {
      const float4* gp =
          (const float4*)(X + (size_t)(blockM + arow) * HIDDEN + k0 + acs);
      float4 a0 = gp[0], a1 = gp[1], a2 = gp[2], a3 = gp[3];
      *(h8*)(&As[arow * LDG + acs])     = pack8(a0, a1);
      *(h8*)(&As[arow * LDG + acs + 8]) = pack8(a2, a3);
    }
    // ---- stage B tile: 32x128 fp32 -> f16, transposed to [n][k] --------
    {
      const float4* gp =
          (const float4*)(W + (size_t)(k0 + bkr) * HIDDEN + blockN + bns);
      float4 b0 = gp[0], b1 = gp[1], b2 = gp[2], b3 = gp[3];
      float v[16] = {b0.x, b0.y, b0.z, b0.w, b1.x, b1.y, b1.z, b1.w,
                     b2.x, b2.y, b2.z, b2.w, b3.x, b3.y, b3.z, b3.w};
#pragma unroll
      for (int j = 0; j < 16; ++j)
        Bs[(bns + j) * LDG + bkr] = (_Float16)v[j];
    }
    if (k0 + 32 < HIDDEN) {   // pull next K-slab toward L2/WGP$
      __builtin_prefetch(X + (size_t)(blockM + arow) * HIDDEN + k0 + 32 + acs, 0, 1);
      __builtin_prefetch(W + (size_t)(k0 + 32 + bkr) * HIDDEN + blockN + bns, 0, 1);
    }
    __syncthreads();

    h16 a0 = ldsA(As, waveM, lane);
    h16 a1 = ldsA(As, waveM + 16, lane);
#pragma unroll
    for (int in = 0; in < 4; ++in) {
      h16 bf = ldsB(Bs, waveN + in * 16, lane);
      acc[0][in] = wmma_f16(a0, bf, acc[0][in]);
      acc[1][in] = wmma_f16(a1, bf, acc[1][in]);
    }
    __syncthreads();
  }

  // ---- epilogue: bias (+ scale / layout change) ------------------------
#pragma unroll
  for (int im = 0; im < 2; ++im) {
#pragma unroll
    for (int in = 0; in < 4; ++in) {
      const int n = blockN + waveN + in * 16 + ln;
      const float bv = bias[n];
#pragma unroll
      for (int r = 0; r < 8; ++r) {
        const int m = blockM + waveM + im * 16 + 8 * g + r;
        const float v = acc[im][in][r] + bv;
        if (MODE == 0) {
          const int hh = n >> 6, d = n & 63;
          const int bb = m >> 12, ss = m & (SEQ - 1);
          ((_Float16*)outp)[((size_t)((bb * HEADS + hh) * SEQ + ss) << 6) + d] =
              (_Float16)(v * scale);
        } else {
          ((float*)outp)[(size_t)m * HIDDEN + n] = v;
        }
      }
    }
  }
}

// ---------------------------------------------------------------------------
// Flash attention: grid (S/64, B*HEADS), 128 threads = 4 waves.
// Wave w owns 16 query rows; loop over key chunks of 64.
//   scores: A = Q frag (global, resident), B = K^T frag (global, contiguous)
//   PV:     A = P (f16 via per-wave LDS),  B = V^T (block-staged LDS)
// Online softmax in fp32 with 16-lane __shfl_xor reductions.
// ---------------------------------------------------------------------------
__global__ __launch_bounds__(128) void attn_kernel(
    const _Float16* __restrict__ Qw, const _Float16* __restrict__ Kw,
    const _Float16* __restrict__ Vw, const float* __restrict__ mask,
    float* __restrict__ ctx) {
  __shared__ _Float16 Vt[64 * LDV];        // V^T for current key chunk: [d][key]
  __shared__ _Float16 Plds[4 * 16 * LDV];  // per-wave P tile: [qrow][key]

  const int tid = threadIdx.x, wave = tid >> 5, lane = tid & 31;
  const int g = lane >> 4, ln = lane & 15;
  const int bh = blockIdx.y, b = bh / HEADS, h = bh - b * HEADS;
  const int qBase = blockIdx.x * 64;

  const _Float16* Q = Qw + (size_t)bh * SEQ * 64;
  const _Float16* K = Kw + (size_t)bh * SEQ * 64;
  const _Float16* V = Vw + (size_t)bh * SEQ * 64;
  const float* mb = mask + (size_t)b * SEQ * SEQ;
  _Float16* Pw = Plds + wave * 16 * LDV;

  // Q fragments (head_dim = 64 -> 2 WMMA K-chunks), resident in VGPRs.
  h16 qf[2];
  {
    const _Float16* p = Q + (size_t)(qBase + wave * 16 + ln) * 64;
#pragma unroll
    for (int kc = 0; kc < 2; ++kc)
      qf[kc] = cat8(*(const h8*)(p + kc * 32 + 8 * g),
                    *(const h8*)(p + kc * 32 + 8 * g + 16));
  }

  float mx[8], lsum[8];
  f8 o[4] = {};
#pragma unroll
  for (int r = 0; r < 8; ++r) { mx[r] = -3.0e38f; lsum[r] = 0.f; }

  for (int j = 0; j < SEQ; j += 64) {
    // ---- stage V^T chunk (64 keys x 64 d -> [d][key]) ------------------
    {
      const int key = tid & 63, dh = (tid >> 6) * 32;
      const h8* vp = (const h8*)(V + (size_t)(j + key) * 64 + dh);
      h8 v0 = vp[0], v1 = vp[1], v2 = vp[2], v3 = vp[3];
#pragma unroll
      for (int i = 0; i < 8; ++i) {
        Vt[(dh + i)      * LDV + key] = v0[i];
        Vt[(dh + 8 + i)  * LDV + key] = v1[i];
        Vt[(dh + 16 + i) * LDV + key] = v2[i];
        Vt[(dh + 24 + i) * LDV + key] = v3[i];
      }
    }
    __syncthreads();

    // ---- scores S = Q K^T (+ mask); Q already pre-scaled by 1/8 --------
    f8 sc[4];
#pragma unroll
    for (int nt = 0; nt < 4; ++nt) {
      f8 s = {};
#pragma unroll
      for (int kc = 0; kc < 2; ++kc) {
        const _Float16* kp = K + (size_t)(j + nt * 16 + ln) * 64 + kc * 32 + 8 * g;
        h16 kf = cat8(*(const h8*)kp, *(const h8*)(kp + 16));
        s = wmma_f16(qf[kc], kf, s);
      }
#pragma unroll
      for (int r = 0; r < 8; ++r)
        s[r] += mb[(size_t)(qBase + wave * 16 + 8 * g + r) * SEQ + j + nt * 16 + ln];
      sc[nt] = s;
    }

    // ---- online softmax -------------------------------------------------
    float nm[8], scf[8], rs[8];
#pragma unroll
    for (int r = 0; r < 8; ++r) {
      nm[r] = mx[r];
#pragma unroll
      for (int nt = 0; nt < 4; ++nt) nm[r] = fmaxf(nm[r], sc[nt][r]);
      nm[r] = fmaxf(nm[r], __shfl_xor(nm[r], 1, 16));
      nm[r] = fmaxf(nm[r], __shfl_xor(nm[r], 2, 16));
      nm[r] = fmaxf(nm[r], __shfl_xor(nm[r], 4, 16));
      nm[r] = fmaxf(nm[r], __shfl_xor(nm[r], 8, 16));
      scf[r] = __expf(mx[r] - nm[r]);
      mx[r] = nm[r];
      rs[r] = 0.f;
    }
#pragma unroll
    for (int nt = 0; nt < 4; ++nt) {
#pragma unroll
      for (int r = 0; r < 8; ++r) {
        const float pv = __expf(sc[nt][r] - mx[r]);
        rs[r] += pv;
        Pw[(8 * g + r) * LDV + nt * 16 + ln] = (_Float16)pv;  // P -> f16 tile
      }
    }
#pragma unroll
    for (int r = 0; r < 8; ++r) {
      rs[r] += __shfl_xor(rs[r], 1, 16);
      rs[r] += __shfl_xor(rs[r], 2, 16);
      rs[r] += __shfl_xor(rs[r], 4, 16);
      rs[r] += __shfl_xor(rs[r], 8, 16);
      lsum[r] = lsum[r] * scf[r] + rs[r];
    }
#pragma unroll
    for (int dt = 0; dt < 4; ++dt)
#pragma unroll
      for (int r = 0; r < 8; ++r) o[dt][r] *= scf[r];

    // ---- O += P V (same-wave LDS ops are in-order) ----------------------
#pragma unroll
    for (int kc = 0; kc < 2; ++kc) {
      const _Float16* pp = Pw + ln * LDV + kc * 32 + 8 * g;
      h16 pf = cat8(*(const h8*)pp, *(const h8*)(pp + 16));
#pragma unroll
      for (int dt = 0; dt < 4; ++dt) {
        const _Float16* vp2 = Vt + (dt * 16 + ln) * LDV + kc * 32 + 8 * g;
        h16 vf = cat8(*(const h8*)vp2, *(const h8*)(vp2 + 16));
        o[dt] = wmma_f16(pf, vf, o[dt]);
      }
    }
    __syncthreads();   // protect Vt before next chunk's restage
  }

  // ---- normalize and write ctx in [b, s, h*64+d] fp32 --------------------
#pragma unroll
  for (int dt = 0; dt < 4; ++dt) {
#pragma unroll
    for (int r = 0; r < 8; ++r) {
      const int qrow = qBase + wave * 16 + 8 * g + r;
      ctx[(size_t)(b * SEQ + qrow) * HIDDEN + h * 64 + dt * 16 + ln] =
          o[dt][r] / lsum[r];
    }
  }
}

// ---------------------------------------------------------------------------
extern "C" void kernel_launch(void* const* d_in, const int* in_sizes, int n_in,
                              void* d_out, int out_size, void* d_ws, size_t ws_size,
                              hipStream_t stream) {
  (void)in_sizes; (void)n_in; (void)out_size;

  const float* key   = (const float*)d_in[0];
  const float* value = (const float*)d_in[1];
  const float* query = (const float*)d_in[2];
  const float* mask  = (const float*)d_in[3];
  const float* Wq = (const float*)d_in[4];
  const float* bq = (const float*)d_in[5];
  const float* Wk = (const float*)d_in[6];
  const float* bk = (const float*)d_in[7];
  const float* Wv = (const float*)d_in[8];
  const float* bv = (const float*)d_in[9];
  const float* Wo = (const float*)d_in[10];
  const float* bo = (const float*)d_in[11];

  const size_t PE = (size_t)BATCH * HEADS * SEQ * 64;  // 6291456 f16 elems
  if (ws_size < PE * 2 * 3 + (size_t)ROWS * HIDDEN * 4) return;

  _Float16* Qh = (_Float16*)d_ws;
  _Float16* Kh = Qh + PE;
  _Float16* Vh = Kh + PE;
  float*    ctx = (float*)(Vh + PE);

  dim3 ggrid(HIDDEN / 128, ROWS / 128);   // (6, 64)
  // Q gets the 1/sqrt(HEAD_DIM) score scale folded in.
  gemm_kernel<0><<<ggrid, 256, 0, stream>>>(query, Wq, bq, (void*)Qh, 0.125f);
  gemm_kernel<0><<<ggrid, 256, 0, stream>>>(key,   Wk, bk, (void*)Kh, 1.0f);
  gemm_kernel<0><<<ggrid, 256, 0, stream>>>(value, Wv, bv, (void*)Vh, 1.0f);

  attn_kernel<<<dim3(SEQ / 64, BATCH * HEADS), 128, 0, stream>>>(Qh, Kh, Vh,
                                                                 mask, ctx);

  gemm_kernel<1><<<ggrid, 256, 0, stream>>>(ctx, Wo, bo, d_out, 1.0f);
}
